// SharedKVDepthMemoryDualQAttention_58222576664897
// MI455X (gfx1250) — matrix operand order repaired and checked by reading, not compiled, not run on hardware
//
#include <hip/hip_runtime.h>
#include <hip/hip_bf16.h>

// B=4, S=1024, D=1024, H=16, L=12, HD=64
// out tuple: out (4M f32) | k_col (4M f32) | v_col (4M f32)

typedef __bf16 bf16;
typedef __attribute__((ext_vector_type(16))) __bf16 v16bf;
typedef __attribute__((ext_vector_type(8)))  __bf16 bf16x8;
typedef __attribute__((ext_vector_type(4)))  __bf16 bf16x4;
typedef __attribute__((ext_vector_type(8)))  float  v8f;
typedef __attribute__((ext_vector_type(4)))  float  f32x4;

#define WMMA_BF16(A, Bm, C) \
  __builtin_amdgcn_wmma_f32_16x16x32_bf16(false, (A), false, (Bm), (short)0, (C), false, false)

static __device__ __forceinline__ bf16 f2bf(float f) {
  union { float f; unsigned u; } x; x.f = f;
  unsigned r = x.u + 0x7FFFu + ((x.u >> 16) & 1u);   // round-to-nearest-even
  unsigned short h = (unsigned short)(r >> 16);
  return __builtin_bit_cast(bf16, h);
}

// ---------------- CDNA5 async global->LDS copy (ASYNCcnt path) ----------------
static __device__ __forceinline__ unsigned lds_off(const void* p) {
  return (unsigned)(size_t)p;   // generic shared addr: low 32 bits = LDS offset
}
static __device__ __forceinline__ void async_g2l_b128(unsigned lds, const void* g) {
  unsigned long long ga = (unsigned long long)(size_t)g;
  asm volatile("global_load_async_to_lds_b128 %0, %1, off" :: "v"(lds), "v"(ga) : "memory");
}
#define WAIT_ASYNC(n) asm volatile("s_wait_asynccnt " #n ::: "memory")

// ---- WMMA fragment loaders (CDNA5 wave32 layouts, CDNA5 ISA) ----
// A (16x32 bf16): row = lane%16; halfwords 0..7 -> k = 8*half.., 8..15 -> k = 16+8*half..
static __device__ __forceinline__ v16bf load_a_frag(const bf16* base, int ld) {
  int lane = threadIdx.x & 31;
  const bf16* p = base + (size_t)(lane & 15) * ld + ((lane >> 4) * 8);
  bf16x8 lo = *(const bf16x8*)p;
  bf16x8 hi = *(const bf16x8*)(p + 16);
  v16bf a;
#pragma unroll
  for (int i = 0; i < 8; ++i) { a[i] = lo[i]; a[8 + i] = hi[i]; }
  return a;
}
// B (32x16 bf16): col = lane%16; halfwords 0..7 -> k = 8*half.., 8..15 -> k = 16+8*half.. base = &Bt[col0][k0],
// Bt stored col-major (row stride ld) -> two 16B reads per lane, starting 32B apart.
static __device__ __forceinline__ v16bf load_b_frag(const bf16* base, int ld) {
  int lane = threadIdx.x & 31;
  const bf16* p = base + (size_t)(lane & 15) * ld + ((lane >> 4) * 8);
  bf16x8 lo = *(const bf16x8*)p;
  bf16x8 hi = *(const bf16x8*)(p + 16);
  v16bf b;
#pragma unroll
  for (int i = 0; i < 8; ++i) { b[i] = lo[i]; b[8 + i] = hi[i]; }
  return b;
}

// -------------------------- cast x -> bf16 --------------------------
__global__ __launch_bounds__(256) void cast_bf16_kernel(const float* __restrict__ src,
                                                        bf16* __restrict__ dst, int n) {
  int i = (blockIdx.x * 256 + threadIdx.x) * 4;
  if (i + 3 < n) {
    f32x4 v = *(const f32x4*)&src[i];
    bf16x4 o;
#pragma unroll
    for (int j = 0; j < 4; ++j) o[j] = f2bf(v[j]);
    *(bf16x4*)&dst[i] = o;
  }
}

// ---------- cast+transpose weights: src (K x N) f32 -> dst (N x K) bf16 ----------
__global__ __launch_bounds__(256) void transcast_kernel(const float* __restrict__ src,
                                                        bf16* __restrict__ dst,
                                                        int K, int N) {
  __shared__ bf16 tile[32][33];
  const int bk = blockIdx.y * 32, bn = blockIdx.x * 32;
  const int tx = threadIdx.x & 31, ty = threadIdx.x >> 5;   // 32 x 8
#pragma unroll
  for (int i = 0; i < 32; i += 8)
    tile[tx][ty + i] = f2bf(src[(size_t)(bk + ty + i) * N + bn + tx]);
  __syncthreads();
#pragma unroll
  for (int i = 0; i < 32; i += 8)
    dst[(size_t)(bn + ty + i) * K + bk + tx] = tile[ty + i][tx];
}

// -------- double-buffered async tile stage: A rows from xbf, B rows from wT --------
#define STAGE_TILE(buf, kk)                                                          \
  do {                                                                               \
    _Pragma("unroll")                                                                \
    for (int i_ = 0; i_ < 2; ++i_) {                                                 \
      int c_ = tid * 2 + i_, r_ = c_ >> 2, ko_ = (c_ & 3) * 8;                       \
      async_g2l_b128(lds_off(&As[buf][r_][ko_]),                                     \
                     &aptr[(size_t)(bm + r_) * 1024 + (kk) + ko_]);                  \
      async_g2l_b128(lds_off(&Bt[buf][r_][ko_]),                                     \
                     &wT[(size_t)(bn + r_) * 1024 + (kk) + ko_]);                    \
    }                                                                                \
  } while (0)

// ---------------- fused QKV + k_col + v_col projection GEMM ----------------
// C(4096 x 5120) = xbf(4096x1024) @ wcatT^T + bias   (wcatT is (5120 x 1024) bf16)
__global__ __launch_bounds__(128) void gemm_qkv_kernel(
    const bf16* __restrict__ aptr, const bf16* __restrict__ wT,
    const float* __restrict__ b_qkv, const float* __restrict__ b_k,
    const float* __restrict__ b_v,
    bf16* __restrict__ qbf, bf16* __restrict__ kbf, bf16* __restrict__ vbf,
    float* __restrict__ kcol_out, float* __restrict__ vcol_out) {
  __shared__ bf16 As[2][64][40];
  __shared__ bf16 Bt[2][64][40];
  const int tid = threadIdx.x, lane = tid & 31, wave = tid >> 5;
  const int half = lane >> 4, col16 = lane & 15;
  const int bm = blockIdx.y * 64;
  const int bn = blockIdx.x * 64;

  int dest; const float* bias;
  if (bn < 3072)      { bias = b_qkv + bn;        dest = 0; }
  else if (bn < 4096) { bias = b_k + (bn - 3072); dest = 1; }
  else                { bias = b_v + (bn - 4096); dest = 2; }

  v8f acc[4];
#pragma unroll
  for (int i = 0; i < 4; ++i) acc[i] = (v8f){0.f,0.f,0.f,0.f,0.f,0.f,0.f,0.f};

  STAGE_TILE(0, 0);
  int cur = 0;
  for (int k0 = 0; k0 < 1024; k0 += 32) {
    if (k0 + 32 < 1024) {
      STAGE_TILE(1 - cur, k0 + 32);
      WAIT_ASYNC(0x4);         // in-order: <=4 outstanding => current tile landed
    } else {
      WAIT_ASYNC(0x0);
    }
    __syncthreads();
    v16bf af = load_a_frag(&As[cur][wave * 16][0], 40);
#pragma unroll
    for (int nt = 0; nt < 4; ++nt) {
      v16bf bfr = load_b_frag(&Bt[cur][nt * 16][0], 40);
      acc[nt] = WMMA_BF16(af, bfr, acc[nt]);
    }
    __syncthreads();           // protect buf before it is refilled next iter
    cur ^= 1;
  }

#pragma unroll
  for (int nt = 0; nt < 4; ++nt) {
    int n_loc = nt * 16 + col16;
    float bv = bias[n_loc];
#pragma unroll
    for (int r = 0; r < 8; ++r) {
      int m = bm + wave * 16 + half * 8 + r;      // C layout: row = r + 8*half
      float val = acc[nt][r] + bv;
      int b = m >> 10, s = m & 1023;
      if (dest == 0) {
        int n = bn + n_loc; int j = n >> 10, d = n & 1023, h = d >> 6, f = d & 63;
        bf16* dst = (j == 0) ? qbf : (j == 1 ? kbf : vbf);
        dst[((((size_t)b * 16 + h) * 1024) + s) * 64 + f] = f2bf(val);
      } else {
        int d = (bn + n_loc) - (dest == 1 ? 3072 : 4096); int h = d >> 6, f = d & 63;
        float* dst = (dest == 1) ? kcol_out : vcol_out;
        dst[((((size_t)b * 16 + h) * 1024) + s) * 64 + f] = val;
      }
    }
  }
}

// --------------------------- output projection GEMM ---------------------------
__global__ __launch_bounds__(128) void gemm_out_kernel(
    const bf16* __restrict__ aptr, const bf16* __restrict__ wT,
    const float* __restrict__ bias, float* __restrict__ cout) {
  __shared__ bf16 As[2][64][40];
  __shared__ bf16 Bt[2][64][40];
  const int tid = threadIdx.x, lane = tid & 31, wave = tid >> 5;
  const int half = lane >> 4, col16 = lane & 15;
  const int bm = blockIdx.y * 64, bn = blockIdx.x * 64;

  v8f acc[4];
#pragma unroll
  for (int i = 0; i < 4; ++i) acc[i] = (v8f){0.f,0.f,0.f,0.f,0.f,0.f,0.f,0.f};

  STAGE_TILE(0, 0);
  int cur = 0;
  for (int k0 = 0; k0 < 1024; k0 += 32) {
    if (k0 + 32 < 1024) {
      STAGE_TILE(1 - cur, k0 + 32);
      WAIT_ASYNC(0x4);
    } else {
      WAIT_ASYNC(0x0);
    }
    __syncthreads();
    v16bf af = load_a_frag(&As[cur][wave * 16][0], 40);
#pragma unroll
    for (int nt = 0; nt < 4; ++nt) {
      v16bf bfr = load_b_frag(&Bt[cur][nt * 16][0], 40);
      acc[nt] = WMMA_BF16(af, bfr, acc[nt]);
    }
    __syncthreads();
    cur ^= 1;
  }

#pragma unroll
  for (int nt = 0; nt < 4; ++nt) {
    int n = bn + nt * 16 + col16;
    float bv = bias[n];
#pragma unroll
    for (int r = 0; r < 8; ++r) {
      int m = bm + wave * 16 + half * 8 + r;
      cout[(size_t)m * 1024 + n] = acc[nt][r] + bv;
    }
  }
}

// ------------------------------ flash attention ------------------------------
// one wave per (b,h,16-query tile); 32-key steps; + L=12 per-row memory slots
__global__ __launch_bounds__(256) void attn_kernel(
    const bf16* __restrict__ qbf, const bf16* __restrict__ kbf, const bf16* __restrict__ vbf,
    const float* __restrict__ past_k, const float* __restrict__ past_v,
    bf16* __restrict__ ctxbf) {
  __shared__ bf16 lds[8 * 3200];
  const int tid = threadIdx.x, lane = tid & 31, wave = tid >> 5;
  const int half = lane >> 4, col16 = lane & 15;
  bf16* PT = &lds[wave * 3200];            // 16 x 40 (P tile, C->A transpose)
  bf16* VT = &lds[wave * 3200 + 640];      // 64 x 40 (V tile transposed: [feat][key])

  int task = blockIdx.x * 8 + wave;        // 4096 tasks
  int qt = task & 63, h = (task >> 6) & 15, b = task >> 10;
  int qbase = qt * 16;

  size_t hoff = ((size_t)b * 16 + h) * 1024 * 64;
  const bf16* qp = qbf + hoff;
  const bf16* kp = kbf + hoff;
  const bf16* vp = vbf + hoff;
  size_t poff = ((size_t)b * 16 + h) * 1024 * 12 * 64;
  const float* pkp = past_k + poff;
  const float* pvp = past_v + poff;

  const float SCALE = 0.125f, NEGBIG = -1e30f;

  v16bf aq0 = load_a_frag(qp + (size_t)qbase * 64, 64);
  v16bf aq1 = load_a_frag(qp + (size_t)qbase * 64 + 32, 64);

  float mrow[8], lrow[8], al[8];
  v8f accv[4];
#pragma unroll
  for (int r = 0; r < 8; ++r) { mrow[r] = NEGBIG; lrow[r] = 0.f; }
#pragma unroll
  for (int ct = 0; ct < 4; ++ct) accv[ct] = (v8f){0.f,0.f,0.f,0.f,0.f,0.f,0.f,0.f};

  const int ntiles = qbase / 32 + 1;
  for (int t = 0; t < ntiles; ++t) {
    const int keyb = t * 32;
    const bool maskt = (t == ntiles - 1);
    if (!maskt) {   // L2 prefetch of next K/V tile rows (global_prefetch_b8)
      __builtin_prefetch(&kp[(size_t)(keyb + 32 + lane) * 64], 0, 1);
      __builtin_prefetch(&vp[(size_t)(keyb + 32 + lane) * 64], 0, 1);
    }

    // ---- scores: S = Q @ K^T (K^T in B-layout is a contiguous per-lane read) ----
    v8f sc2[2];
#pragma unroll
    for (int s2 = 0; s2 < 2; ++s2) {
      const bf16* kb = kp + (size_t)(keyb + s2 * 16) * 64;
      v16bf b0 = load_b_frag(kb, 64);        // features 0..31
      v16bf b1 = load_b_frag(kb + 32, 64);   // features 32..63
      v8f z = (v8f){0.f,0.f,0.f,0.f,0.f,0.f,0.f,0.f};
      z = WMMA_BF16(aq0, b0, z);
      z = WMMA_BF16(aq1, b1, z);
      sc2[s2] = z;
    }

    float s0v[8], s1v[8];
#pragma unroll
    for (int r = 0; r < 8; ++r) {
      float a0 = sc2[0][r] * SCALE, a1 = sc2[1][r] * SCALE;
      if (maskt) {
        int rg = qbase + half * 8 + r;
        if (keyb + col16 > rg)      a0 = NEGBIG;
        if (keyb + 16 + col16 > rg) a1 = NEGBIG;
      }
      s0v[r] = a0; s1v[r] = a1;
    }

    // ---- online softmax (row reductions across 16-lane groups) ----
#pragma unroll
    for (int r = 0; r < 8; ++r) {
      float mx = fmaxf(s0v[r], s1v[r]);
#pragma unroll
      for (int m = 8; m >= 1; m >>= 1) mx = fmaxf(mx, __shfl_xor(mx, m, 16));
      float mn = fmaxf(mrow[r], mx);
      float a  = __expf(mrow[r] - mn);
      float p0 = __expf(s0v[r] - mn);
      float p1 = __expf(s1v[r] - mn);
      PT[(half * 8 + r) * 40 + col16]      = f2bf(p0);
      PT[(half * 8 + r) * 40 + 16 + col16] = f2bf(p1);
      float ps = p0 + p1;
#pragma unroll
      for (int m = 8; m >= 1; m >>= 1) ps += __shfl_xor(ps, m, 16);
      lrow[r] = lrow[r] * a + ps;
      mrow[r] = mn;
      al[r] = a;
    }
#pragma unroll
    for (int ct = 0; ct < 4; ++ct)
#pragma unroll
      for (int r = 0; r < 8; ++r) accv[ct][r] *= al[r];

    // ---- stage V tile transposed: VT[feat][key], lane = key ----
#pragma unroll
    for (int c = 0; c < 8; ++c) {
      bf16x8 vv = *(const bf16x8*)&vp[(size_t)(keyb + lane) * 64 + c * 8];
#pragma unroll
      for (int j = 0; j < 8; ++j) VT[(c * 8 + j) * 40 + lane] = vv[j];
    }
    asm volatile("s_wait_dscnt 0x0" ::: "memory");   // intra-wave LDS RAW fence

    // ---- ctx += P @ V ----
    v16bf ap = load_a_frag(PT, 40);
#pragma unroll
    for (int ct = 0; ct < 4; ++ct) {
      v16bf bv = load_b_frag(VT + ct * 16 * 40, 40);
      accv[ct] = WMMA_BF16(ap, bv, accv[ct]);
    }
  }

  // ---- per-position memory slots: 12 keys/values private to each query row ----
  float sc[12];
  {
    const bf16*  qrow  = qp  + (size_t)(qbase + col16) * 64 + half * 32;
    const float* pkrow = pkp + (size_t)(qbase + col16) * 12 * 64 + half * 32;
#pragma unroll
    for (int l = 0; l < 12; ++l) {
      float d = 0.f;
#pragma unroll
      for (int j = 0; j < 32; ++j) d += (float)qrow[j] * pkrow[l * 64 + j];
      d += __shfl_xor(d, 16, 32);      // combine feature halves
      sc[l] = d * SCALE;
    }
  }
  float mx12 = sc[0];
#pragma unroll
  for (int l = 1; l < 12; ++l) mx12 = fmaxf(mx12, sc[l]);

#pragma unroll
  for (int r = 0; r < 8; ++r) {
    int rg = half * 8 + r;                       // row of this lane in C layout
    float mxr = __shfl(mx12, rg, 32);            // lane 'rg' holds row rg's stats
    float mn  = fmaxf(mrow[r], mxr);
    float a   = __expf(mrow[r] - mn);
#pragma unroll
    for (int ct = 0; ct < 4; ++ct) accv[ct][r] *= a;
    float sumE = 0.f;
    const float* pvrow = pvp + (size_t)(qbase + rg) * 12 * 64;
#pragma unroll
    for (int l = 0; l < 12; ++l) {
      float w = __shfl(sc[l], rg, 32);
      float e = __expf(w - mn);
      sumE += e;
#pragma unroll
      for (int ct = 0; ct < 4; ++ct)
        accv[ct][r] += e * pvrow[l * 64 + ct * 16 + col16];
    }
    lrow[r] = lrow[r] * a + sumE;
    mrow[r] = mn;
  }

  // ---- normalize + store ctx as bf16 in (B, S, H*HD) layout for output GEMM ----
#pragma unroll
  for (int r = 0; r < 8; ++r) {
    float inv = 1.f / lrow[r];
    int rg = half * 8 + r;
    size_t rowaddr = ((size_t)b * 1024 + qbase + rg) * 1024 + (size_t)h * 64;
#pragma unroll
    for (int ct = 0; ct < 4; ++ct)
      ctxbf[rowaddr + ct * 16 + col16] = f2bf(accv[ct][r] * inv);
  }
}

// --------------------------------- launcher ---------------------------------
extern "C" void kernel_launch(void* const* d_in, const int* in_sizes, int n_in,
                              void* d_out, int out_size, void* d_ws, size_t ws_size,
                              hipStream_t stream) {
  const float* x      = (const float*)d_in[0];
  const float* past_k = (const float*)d_in[1];
  const float* past_v = (const float*)d_in[2];
  const float* w_qkv  = (const float*)d_in[3];
  const float* b_qkv  = (const float*)d_in[4];
  const float* w_k    = (const float*)d_in[5];
  const float* b_k    = (const float*)d_in[6];
  const float* w_v    = (const float*)d_in[7];
  const float* b_v    = (const float*)d_in[8];
  const float* w_o    = (const float*)d_in[9];
  const float* b_o    = (const float*)d_in[10];

  float* out  = (float*)d_out;            // (B,S,D)
  float* kcol = out + 4194304;            // (B,H,S,HD)
  float* vcol = out + 8388608;            // (B,H,S,HD)

  char* ws = (char*)d_ws;                 // ~52 MB used
  bf16* xbf   = (bf16*)(ws + 0);          // 8 MB
  bf16* qbf   = (bf16*)(ws + 8388608);
  bf16* kbf   = (bf16*)(ws + 16777216);
  bf16* vbf   = (bf16*)(ws + 25165824);
  bf16* ctxbf = (bf16*)(ws + 33554432);
  bf16* wcatT = (bf16*)(ws + 41943040);   // (5120 x 1024) bf16 = 10 MB
  bf16* woT   = (bf16*)(ws + 52428800);   // (1024 x 1024) bf16 = 2 MB

  cast_bf16_kernel<<<4096, 256, 0, stream>>>(x, xbf, 4194304);
  transcast_kernel<<<dim3(96, 32), 256, 0, stream>>>(w_qkv, wcatT, 1024, 3072);
  transcast_kernel<<<dim3(32, 32), 256, 0, stream>>>(w_k, wcatT + 3145728, 1024, 1024);
  transcast_kernel<<<dim3(32, 32), 256, 0, stream>>>(w_v, wcatT + 4194304, 1024, 1024);
  transcast_kernel<<<dim3(32, 32), 256, 0, stream>>>(w_o, woT, 1024, 1024);

  gemm_qkv_kernel<<<dim3(80, 64), 128, 0, stream>>>(
      xbf, wcatT, b_qkv, b_k, b_v, qbf, kbf, vbf, kcol, vcol);
  attn_kernel<<<512, 256, 0, stream>>>(qbf, kbf, vbf, past_k, past_v, ctxbf);
  gemm_out_kernel<<<dim3(16, 64), 128, 0, stream>>>(ctxbf, woT, b_o, out);
}
